// TRANSNET_32856499814772
// MI455X (gfx1250) — matrix operand order repaired and checked by reading, not compiled, not run on hardware
//
#include <hip/hip_runtime.h>

// ---------------- problem constants ----------------
#define V_  50000
#define D_  64
#define F_  100
#define K_  3
#define L_  256
#define R_  10
#define ID_ 32
#define B_  128

#define FT_ 7    // F tiles of 16 (100 -> 112)
#define LT_ 16   // L tiles of 16
#define KS_ 6    // K steps of 32 (192 = 6*32)
#define ROWP_ 72 // padded LDS row length in halves (64 + 8 pad, anti-bank-conflict)

typedef _Float16 v16h __attribute__((ext_vector_type(16)));
typedef _Float16 v8h  __attribute__((ext_vector_type(8)));
typedef float    v8f  __attribute__((ext_vector_type(8)));

union V16 { v16h v; v8h h[2]; };

// ---------------- workspace layout (bytes) ----------------
// reviews : B*L int                    @ 0
// ul      : B*R*ID float               @ WS_UL
// il      : B*R*ID float               @ WS_IL
// Aprep   : 3 * FT*KS*32*16 _Float16   @ WS_AP
#define WS_REV 0
#define WS_UL  (WS_REV + B_*L_*4)
#define WS_IL  (WS_UL + B_*R_*ID_*4)
#define WS_AP  (WS_IL + B_*R_*ID_*4)
#define APREP_PER_CONV (FT_*KS_*32)   // lanes per conv; each lane owns 16 halves

// =====================================================================
// Repack conv weights [F,D,K] f32 -> f16 A-fragments, per-lane order.
// K ordering: c = kker*64 + d  (so B fragments are contiguous in LDS).
// =====================================================================
__global__ void prep_weights(const float* __restrict__ cu,
                             const float* __restrict__ ci,
                             const float* __restrict__ tc,
                             _Float16* __restrict__ out) {
    int tid = blockIdx.x * blockDim.x + threadIdx.x;
    if (tid >= 3 * APREP_PER_CONV) return;
    int conv = tid / APREP_PER_CONV;
    int rem  = tid % APREP_PER_CONV;
    int t    = rem / (KS_ * 32);
    int rem2 = rem % (KS_ * 32);
    int s    = rem2 / 32;
    int lane = rem2 % 32;
    const float* Wc = (conv == 0) ? cu : ((conv == 1) ? ci : tc);
    _Float16* o = out + (size_t)tid * 16;

    int m  = lane & 15;
    int kb = (lane & 16) ? 16 : 0;
    int f  = t * 16 + m;
#pragma unroll
    for (int i = 0; i < 16; ++i) {
        int j = i >> 1, p = i & 1;
        int kk = kb + 2 * j + p;
        int c  = s * 32 + kk;              // 0..191
        int d  = c & 63;
        int kker = c >> 6;                 // 0..2
        float val = (f < F_) ? Wc[((size_t)f * D_ + d) * K_ + kker] : 0.f;
        o[i] = (_Float16)val;
    }
}

// =====================================================================
// Per-sample review selection (first-match semantics like argmax).
// =====================================================================
__global__ __launch_bounds__(L_) void select_reviews(
    const int* __restrict__ u2i, const int* __restrict__ i2u,
    const int* __restrict__ uids, const int* __restrict__ iids,
    const int* __restrict__ ur, const int* __restrict__ ir,
    const int* __restrict__ rr, int* __restrict__ out) {
    int b = blockIdx.x;
    __shared__ const int* srcp;
    if (threadIdx.x == 0) {
        int j1 = -1, j2 = -1;
        for (int r = R_ - 1; r >= 0; --r) {
            if (u2i[b * R_ + r] == iids[b]) j1 = r;
            if (i2u[b * R_ + r] == uids[b]) j2 = r;
        }
        const int* p;
        if (j1 >= 0)      p = ur + ((size_t)b * R_ + j1) * L_;
        else if (j2 >= 0) p = ir + ((size_t)b * R_ + j2) * L_;
        else              p = rr + (size_t)b * L_;
        srcp = p;
    }
    __syncthreads();
    out[(size_t)b * L_ + threadIdx.x] = srcp[threadIdx.x];
}

// =====================================================================
// CNN: conv1d(K=3,pad=1) as WMMA GEMM [F=112pad x 192] x [192 x 256],
// fused relu+maxpool(L) + linear(ID,F) + tanh.
// One workgroup (7 waves) per sequence.
// =====================================================================
__global__ __launch_bounds__(FT_ * 32) void cnn_kernel(
    const int*   __restrict__ tokens,  // [n_seq, L]
    const float* __restrict__ emb,     // [V, D]
    const _Float16* __restrict__ Aprep,// [FT*KS*32*16]
    const float* __restrict__ bc,      // [F]
    const float* __restrict__ Wl,      // [ID, F]
    const float* __restrict__ bl,      // [ID]
    float* __restrict__ outp)          // [n_seq, ID]
{
    __shared__ _Float16 embS[(L_ + 2) * ROWP_]; // rows: 0 = x[-1] pad, 257 = x[256] pad
    __shared__ int   ids[L_];
    __shared__ float colmax[FT_ * 16];

    const int seq  = blockIdx.x;
    const int tid  = threadIdx.x;
    const int lane = tid & 31;
    const int wave = tid >> 5;
    const int NT   = FT_ * 32;

    // token ids
    for (int i = tid; i < L_; i += NT) ids[i] = tokens[(size_t)seq * L_ + i];
    // zero conv-padding rows
    for (int i = tid; i < 2 * ROWP_; i += NT)
        embS[(i < ROWP_) ? i : ((L_ + 1) * ROWP_ + i - ROWP_)] = (_Float16)0.f;
    __syncthreads();

    // gather embedding rows, f32 -> f16, vectorized by float4
    for (int i = tid; i < L_ * (D_ / 4); i += NT) {
        int row = i >> 4;       // D/4 = 16 float4 per row
        int c4  = i & 15;
        float4 v = ((const float4*)(emb + (size_t)ids[row] * D_))[c4];
        _Float16* dst = &embS[(row + 1) * ROWP_ + c4 * 4];
        dst[0] = (_Float16)v.x; dst[1] = (_Float16)v.y;
        dst[2] = (_Float16)v.z; dst[3] = (_Float16)v.w;
    }
    __syncthreads();

    // ---- WMMA GEMM + running max-pool ----
    {
        const int t = wave;  // row tile: f = 16t .. 16t+15
        V16 afrag[KS_];
#pragma unroll
        for (int s = 0; s < KS_; ++s) {
            const _Float16* ap = Aprep + ((size_t)(t * KS_ + s) * 32 + lane) * 16;
            afrag[s].h[0] = *(const v8h*)ap;
            afrag[s].h[1] = *(const v8h*)(ap + 8);
        }

        v8f vmax;
#pragma unroll
        for (int r = 0; r < 8; ++r) vmax[r] = -1e30f;

        const int khalf = (lane & 16) ? 16 : 0;
        for (int lt = 0; lt < LT_; ++lt) {
            v8f acc = {};
#pragma unroll
            for (int s = 0; s < KS_; ++s) {
                int kbase = s * 32 + khalf;      // 0..191, step 16
                int kker  = kbase >> 6;          // conv tap 0..2
                int d0    = kbase & 63;          // channel offset (mult of 16)
                int row   = lt * 16 + (lane & 15) + kker;
                const _Float16* bp = &embS[row * ROWP_ + d0];
                V16 bfrag;
                bfrag.h[0] = *(const v8h*)bp;
                bfrag.h[1] = *(const v8h*)(bp + 8);
                acc = __builtin_amdgcn_wmma_f32_16x16x32_f16(
                        false, afrag[s].v, false, bfrag.v,
                        (short)0, acc, false, false);
            }
#pragma unroll
            for (int r = 0; r < 8; ++r) vmax[r] = fmaxf(vmax[r], acc[r]);
        }

        // reduce max over N (16 lanes per half-wave)
#pragma unroll
        for (int off = 1; off < 16; off <<= 1) {
#pragma unroll
            for (int r = 0; r < 8; ++r)
                vmax[r] = fmaxf(vmax[r], __shfl_xor(vmax[r], off, 32));
        }
        if ((lane & 15) == 0) {
            int mbase = t * 16 + ((lane & 16) ? 8 : 0);
#pragma unroll
            for (int r = 0; r < 8; ++r) colmax[mbase + r] = vmax[r];
        }
    }
    __syncthreads();

    // ---- bias + relu + linear(ID,F) + tanh ----
    if (tid < ID_) {
        float acc = bl[tid];
        for (int f = 0; f < F_; ++f) {
            float m = fmaxf(colmax[f] + bc[f], 0.f);
            acc += m * Wl[tid * F_ + f];
        }
        outp[(size_t)seq * ID_ + tid] = tanhf(acc);
    }
}

// =====================================================================
// TransNet: src = tanh(tanh(cat(ul,il) @ W1^T + b1) @ W2^T + b2)
// =====================================================================
__global__ __launch_bounds__(ID_) void transnet(
    const float* __restrict__ ul, const float* __restrict__ il,
    const float* __restrict__ W1, const float* __restrict__ b1,
    const float* __restrict__ W2, const float* __restrict__ b2,
    float* __restrict__ out) {
    int b = blockIdx.x, i = threadIdx.x;  // 32 threads
    __shared__ float h[ID_];
    const int HC = R_ * ID_;              // 320
    float acc = b1[i];
    const float* w = W1 + (size_t)i * (2 * HC);
    const float* u = ul + (size_t)b * HC;
    const float* v = il + (size_t)b * HC;
    for (int c = 0; c < HC; ++c) acc += u[c] * w[c];
    for (int c = 0; c < HC; ++c) acc += v[c] * w[HC + c];
    h[i] = tanhf(acc);
    __syncthreads();
    float acc2 = b2[i];
#pragma unroll
    for (int c = 0; c < ID_; ++c) acc2 += h[c] * W2[i * ID_ + c];
    out[(size_t)b * ID_ + i] = tanhf(acc2);
}

// =====================================================================
extern "C" void kernel_launch(void* const* d_in, const int* in_sizes, int n_in,
                              void* d_out, int out_size, void* d_ws, size_t ws_size,
                              hipStream_t stream) {
    const int*   user_reviews = (const int*)d_in[0];
    const int*   item_reviews = (const int*)d_in[1];
    const int*   uids         = (const int*)d_in[2];
    const int*   iids         = (const int*)d_in[3];
    const int*   user2item    = (const int*)d_in[4];
    const int*   item2user    = (const int*)d_in[5];
    const int*   rand_reviews = (const int*)d_in[6];
    const float* user_emb     = (const float*)d_in[7];
    const float* item_emb     = (const float*)d_in[8];
    const float* cu_Wc = (const float*)d_in[9];
    const float* cu_bc = (const float*)d_in[10];
    const float* cu_Wl = (const float*)d_in[11];
    const float* cu_bl = (const float*)d_in[12];
    const float* ci_Wc = (const float*)d_in[13];
    const float* ci_bc = (const float*)d_in[14];
    const float* ci_Wl = (const float*)d_in[15];
    const float* ci_bl = (const float*)d_in[16];
    const float* t_W1  = (const float*)d_in[17];
    const float* t_b1  = (const float*)d_in[18];
    const float* t_W2  = (const float*)d_in[19];
    const float* t_b2  = (const float*)d_in[20];
    // d_in[21..23] = fs_* (dead code in output)
    const float* te_emb = (const float*)d_in[24];
    const float* tc_Wc  = (const float*)d_in[25];
    const float* tc_bc  = (const float*)d_in[26];
    const float* tc_Wl  = (const float*)d_in[27];
    const float* tc_bl  = (const float*)d_in[28];
    // d_in[29..31] = ft_* (dead code in output)

    char* ws = (char*)d_ws;
    int*       rev_ws = (int*)(ws + WS_REV);
    float*     ul_ws  = (float*)(ws + WS_UL);
    float*     il_ws  = (float*)(ws + WS_IL);
    _Float16*  ap_ws  = (_Float16*)(ws + WS_AP);

    float* src_out = (float*)d_out;            // [B, ID]
    float* tl_out  = src_out + B_ * ID_;       // [B, ID]

    // 1) repack conv weights to f16 WMMA A-fragments
    {
        int total = 3 * APREP_PER_CONV;
        prep_weights<<<(total + 255) / 256, 256, 0, stream>>>(cu_Wc, ci_Wc, tc_Wc, ap_ws);
    }
    // 2) target review selection
    select_reviews<<<B_, L_, 0, stream>>>(user2item, item2user, uids, iids,
                                          user_reviews, item_reviews, rand_reviews, rev_ws);
    // 3) SourceNet CNNs (user / item), 1280 sequences each
    cnn_kernel<<<B_ * R_, FT_ * 32, 0, stream>>>(
        user_reviews, user_emb, ap_ws + (size_t)0 * APREP_PER_CONV * 16,
        cu_bc, cu_Wl, cu_bl, ul_ws);
    cnn_kernel<<<B_ * R_, FT_ * 32, 0, stream>>>(
        item_reviews, item_emb, ap_ws + (size_t)1 * APREP_PER_CONV * 16,
        ci_bc, ci_Wl, ci_bl, il_ws);
    // 4) TargetNet CNN -> tl (second output)
    cnn_kernel<<<B_, FT_ * 32, 0, stream>>>(
        rev_ws, te_emb, ap_ws + (size_t)2 * APREP_PER_CONV * 16,
        tc_bc, tc_Wl, tc_bl, tl_out);
    // 5) TransNet MLP -> src (first output)
    transnet<<<B_, ID_, 0, stream>>>(ul_ws, il_ws, t_W1, t_b1, t_W2, t_b2, src_out);
}
